// Attention_62861141344964
// MI455X (gfx1250) — matrix operand order repaired and checked by reading, not compiled, not run on hardware
//
#include <hip/hip_runtime.h>

// ---------------------------------------------------------------------------
// MI455X (gfx1250) self-attention block, bf16 WMMA + flash softmax.
// ~36 GFLOP, ~30 MB working set -> L2-resident, WMMA-issue/latency limited.
// GEMMs use 32x64 tiles/wave: each B fragment feeds 2 WMMAs (32 FLOP/byte),
// software-pipelined (unroll-2 ping-pong) fragment loads hide L2 latency.
// ---------------------------------------------------------------------------

typedef __attribute__((ext_vector_type(16))) __bf16 v16bf;
typedef __attribute__((ext_vector_type(8)))  float  v8f;
typedef __attribute__((ext_vector_type(2)))  __bf16 bf16x2;

#define DEVINL __device__ __forceinline__

constexpr int NPIX = 48 * 48;      // 2304 pixels per batch
constexpr int NB   = 2;
constexpr int NG   = NB * NPIX;    // 4608 total pixels
constexpr int CDIM = 512;
constexpr int HEADS = 8;
constexpr int DH    = 64;
constexpr int OQKV  = 3 * CDIM;    // 1536

DEVINL v8f v8f_zero() {
  v8f z;
#pragma unroll
  for (int i = 0; i < 8; ++i) z[i] = 0.0f;
  return z;
}

// A fragment, 16x32 (MxK) bf16. Memory: row-major [m][k], row stride ld.
DEVINL v16bf load_a_frag(const __bf16* src, int ld) {
  const int lane = threadIdx.x & 31;
  const int m  = lane & 15;
  const int kh = (lane >> 4) << 3;
  const __bf16* row = src + (size_t)m * ld;
  v16bf a;
#pragma unroll
  for (int t = 0; t < 4; ++t) {
    bf16x2 lo = *(const bf16x2*)(row + kh + 2 * t);
    bf16x2 hi = *(const bf16x2*)(row + 16 + kh + 2 * t);
    a[2 * t]     = lo.x; a[2 * t + 1]     = lo.y;
    a[8 + 2 * t] = hi.x; a[8 + 2 * t + 1] = hi.y;
  }
  return a;
}

// B fragment, 32x16 (KxN) bf16. Memory: N-major [n][k], row stride ld.
DEVINL v16bf load_b_frag(const __bf16* src, int ld) {
  const int lane = threadIdx.x & 31;
  const int n  = lane & 15;
  const int kb = (lane >> 4) << 3;
  const __bf16* col = src + (size_t)n * ld + kb;
  v16bf b;
#pragma unroll
  for (int t = 0; t < 4; ++t) {
    bf16x2 p = *(const bf16x2*)(col + 2 * t);
    bf16x2 q = *(const bf16x2*)(col + 16 + 2 * t);
    b[2 * t] = p.x; b[2 * t + 1] = p.y;
    b[8 + 2 * t] = q.x; b[8 + 2 * t + 1] = q.y;
  }
  return b;
}

DEVINL v8f wmma_bf16(v16bf a, v16bf b, v8f c) {
  return __builtin_amdgcn_wmma_f32_16x16x32_bf16(false, a, false, b, (short)0, c,
                                                 false, false);
}

DEVINL float red16_max(float v) {
#pragma unroll
  for (int m = 1; m <= 8; m <<= 1) v = fmaxf(v, __shfl_xor(v, m, 32));
  return v;
}
DEVINL float red16_sum(float v) {
#pragma unroll
  for (int m = 1; m <= 8; m <<= 1) v += __shfl_xor(v, m, 32);
  return v;
}

// ---------------------------------------------------------------------------
// Kernel 0: convert weights fp32 -> bf16.
// ---------------------------------------------------------------------------
__global__ __launch_bounds__(256) void cvt_weights_kernel(
    const float* __restrict__ wqkv, const float* __restrict__ wout,
    __bf16* __restrict__ wqkv_bf, __bf16* __restrict__ wout_bf) {
  int i = blockIdx.x * 256 + threadIdx.x;
  if (i < OQKV * CDIM) wqkv_bf[i] = (__bf16)wqkv[i];
  if (i < CDIM * CDIM) wout_bf[i] = (__bf16)wout[i];
}

// ---------------------------------------------------------------------------
// Kernel 1: channel RMSNorm. x[b][c][n] -> xnt[b*n][c] (bf16, c-contiguous).
// ---------------------------------------------------------------------------
__global__ __launch_bounds__(256) void rmsnorm_kernel(
    const float* __restrict__ x, const float* __restrict__ g,
    __bf16* __restrict__ xnt) {
  int ng = blockIdx.x * 256 + threadIdx.x;
  if (ng >= NG) return;
  int b = ng / NPIX, n = ng % NPIX;
  const float* base = x + (size_t)b * CDIM * NPIX + n;
  float ss = 0.0f;
  for (int c = 0; c < CDIM; ++c) {
    float v = base[(size_t)c * NPIX];
    ss += v * v;
  }
  float sc = 22.627416997969522f / fmaxf(sqrtf(ss), 1e-12f);  // sqrt(512)/l2
  __bf16* dst = xnt + (size_t)ng * CDIM;
  for (int c = 0; c < CDIM; ++c)
    dst[c] = (__bf16)(base[(size_t)c * NPIX] * sc * g[c]);
}

// ---------------------------------------------------------------------------
// Kernel 2: QKV GEMM, 32(M) x 64(N) tile per wave, pipelined K loop.
// Epilogue scatters q (scaled) -> qt[bh][n][d], k -> kt[bh][n][d],
// v -> vb[bh][d][n] (bf16).
// ---------------------------------------------------------------------------
__global__ __launch_bounds__(256) void qkv_gemm_kernel(
    const __bf16* __restrict__ w, const __bf16* __restrict__ xnt,
    __bf16* __restrict__ qt, __bf16* __restrict__ kt, __bf16* __restrict__ vb) {
  const int wave = threadIdx.x >> 5, lane = threadIdx.x & 31;
  const int tile = blockIdx.x * 8 + wave;        // 48 * 72 = 3456 tiles
  const int mt = tile % (OQKV / 32);
  const int nt = tile / (OQKV / 32);
  const int m0 = mt * 32, n0 = nt * 64;

  const __bf16* arow0 = w + (size_t)m0 * CDIM;
  const __bf16* arow1 = w + (size_t)(m0 + 16) * CDIM;

  v8f acc[2][4];
#pragma unroll
  for (int mi = 0; mi < 2; ++mi)
#pragma unroll
    for (int t = 0; t < 4; ++t) acc[mi][t] = v8f_zero();

  v16bf a_cur[2], b_cur[4];
  a_cur[0] = load_a_frag(arow0, CDIM);
  a_cur[1] = load_a_frag(arow1, CDIM);
#pragma unroll
  for (int t = 0; t < 4; ++t)
    b_cur[t] = load_b_frag(xnt + (size_t)(n0 + 16 * t) * CDIM, CDIM);

#pragma unroll 2
  for (int k0 = 32; k0 < CDIM; k0 += 32) {
    if (k0 + 32 < CDIM) {                       // prefetch one step deeper
      __builtin_prefetch(arow0 + k0 + 32, 0, 3);
      __builtin_prefetch(xnt + (size_t)n0 * CDIM + k0 + 32, 0, 3);
    }
    v16bf a_nxt[2], b_nxt[4];
    a_nxt[0] = load_a_frag(arow0 + k0, CDIM);
    a_nxt[1] = load_a_frag(arow1 + k0, CDIM);
#pragma unroll
    for (int t = 0; t < 4; ++t)
      b_nxt[t] = load_b_frag(xnt + (size_t)(n0 + 16 * t) * CDIM + k0, CDIM);
#pragma unroll
    for (int t = 0; t < 4; ++t) {
      acc[0][t] = wmma_bf16(a_cur[0], b_cur[t], acc[0][t]);
      acc[1][t] = wmma_bf16(a_cur[1], b_cur[t], acc[1][t]);
    }
    a_cur[0] = a_nxt[0]; a_cur[1] = a_nxt[1];
#pragma unroll
    for (int t = 0; t < 4; ++t) b_cur[t] = b_nxt[t];
  }
#pragma unroll
  for (int t = 0; t < 4; ++t) {
    acc[0][t] = wmma_bf16(a_cur[0], b_cur[t], acc[0][t]);
    acc[1][t] = wmma_bf16(a_cur[1], b_cur[t], acc[1][t]);
  }

  const int nl = lane & 15, hi = lane >> 4;
#pragma unroll
  for (int mi = 0; mi < 2; ++mi) {
#pragma unroll
    for (int t = 0; t < 4; ++t) {
#pragma unroll
      for (int r = 0; r < 8; ++r) {
        int o   = m0 + mi * 16 + r + 8 * hi;
        int ngl = n0 + 16 * t + nl;
        int b = ngl / NPIX, n = ngl % NPIX;
        float val = acc[mi][t][r];
        if (o < CDIM) {                               // q, fold in dh^-0.5
          int h = o >> 6, d = o & 63;
          qt[(((size_t)(b * HEADS + h)) * NPIX + n) * DH + d] = (__bf16)(val * 0.125f);
        } else if (o < 2 * CDIM) {                    // k
          int oo = o - CDIM; int h = oo >> 6, d = oo & 63;
          kt[(((size_t)(b * HEADS + h)) * NPIX + n) * DH + d] = (__bf16)val;
        } else {                                      // v
          int oo = o - 2 * CDIM; int h = oo >> 6, d = oo & 63;
          vb[(((size_t)(b * HEADS + h)) * DH + d) * NPIX + n] = (__bf16)val;
        }
      }
    }
  }
}

// ---------------------------------------------------------------------------
// Kernel 3: flash attention. One wave per 16-row i-tile of one (b,h)
// (2304 independent wave-tasks; kept at 16 rows for occupancy).
// K/V fragments for the NEXT j-step are issued before the current step's
// WMMAs + softmax + LDS transpose; unroll-2 removes buffer-rotation moves.
// ---------------------------------------------------------------------------
DEVINL void attn_step(v16bf aq0, v16bf aq1, const v16bf kb[4], const v16bf vf[4],
                      v8f acc[4], float mrow[8], float lrow[8],
                      __bf16* pt, int nl, int hi) {
  v8f s0 = v8f_zero(), s1 = v8f_zero();
  s0 = wmma_bf16(aq0, kb[0], s0);
  s0 = wmma_bf16(aq1, kb[1], s0);
  s1 = wmma_bf16(aq0, kb[2], s1);
  s1 = wmma_bf16(aq1, kb[3], s1);

#pragma unroll
  for (int r = 0; r < 8; ++r) {
    float tm = red16_max(fmaxf(s0[r], s1[r]));
    float nm = fmaxf(mrow[r], tm);
    float p0 = __expf(s0[r] - nm);
    float p1 = __expf(s1[r] - nm);
    float rs = red16_sum(p0 + p1);
    float alpha = __expf(mrow[r] - nm);
    lrow[r] = lrow[r] * alpha + rs;
    mrow[r] = nm;
    s0[r] = p0; s1[r] = p1;
#pragma unroll
    for (int t = 0; t < 4; ++t) acc[t][r] *= alpha;
  }

#pragma unroll
  for (int r = 0; r < 8; ++r) {
    int row = r + 8 * hi;
    pt[row * 32 + nl]      = (__bf16)s0[r];
    pt[row * 32 + 16 + nl] = (__bf16)s1[r];
  }
  asm volatile("s_wait_dscnt 0" ::: "memory");   // same-wave DS RAW fence
  v16bf ap = load_a_frag(pt, 32);
  asm volatile("" ::: "memory");                 // keep later stores below

#pragma unroll
  for (int t = 0; t < 4; ++t) acc[t] = wmma_bf16(ap, vf[t], acc[t]);
}

__global__ __launch_bounds__(128) void attn_kernel(
    const __bf16* __restrict__ qt, const __bf16* __restrict__ kt,
    const __bf16* __restrict__ vb, __bf16* __restrict__ ao) {
  __shared__ __align__(16) __bf16 lds_p[4][16 * 32];
  const int wave = threadIdx.x >> 5;
  const int lane = threadIdx.x & 31;
  const int bh = blockIdx.y;                     // 0..15
  const int itile = blockIdx.x * 4 + wave;       // 0..143
  const int i0 = itile * 16;
  const int b = bh >> 3, h = bh & 7;

  const __bf16* q = qt + (size_t)bh * NPIX * DH;
  const __bf16* k = kt + (size_t)bh * NPIX * DH;
  const __bf16* v = vb + (size_t)bh * DH * NPIX;

  v16bf aq0 = load_a_frag(q + (size_t)i0 * DH, DH);        // K = d 0..31
  v16bf aq1 = load_a_frag(q + (size_t)i0 * DH + 32, DH);   // K = d 32..63

  float mrow[8], lrow[8];
  v8f acc[4];
#pragma unroll
  for (int r = 0; r < 8; ++r) { mrow[r] = -3.0e38f; lrow[r] = 0.0f; }
#pragma unroll
  for (int t = 0; t < 4; ++t) acc[t] = v8f_zero();

  __bf16* pt = lds_p[wave];
  const int nl = lane & 15, hi = lane >> 4;

  // preload fragments for j0 = 0
  v16bf kb[4], vf[4];
  kb[0] = load_b_frag(k, DH);
  kb[1] = load_b_frag(k + 32, DH);
  kb[2] = load_b_frag(k + (size_t)16 * DH, DH);
  kb[3] = load_b_frag(k + (size_t)16 * DH + 32, DH);
#pragma unroll
  for (int t = 0; t < 4; ++t) vf[t] = load_b_frag(v + (size_t)(t * 16) * NPIX, NPIX);

#pragma unroll 2
  for (int j0 = 0; j0 < NPIX - 32; j0 += 32) {
    const int j1 = j0 + 32;
    v16bf nkb[4], nvf[4];
    nkb[0] = load_b_frag(k + (size_t)j1 * DH, DH);
    nkb[1] = load_b_frag(k + (size_t)j1 * DH + 32, DH);
    nkb[2] = load_b_frag(k + (size_t)(j1 + 16) * DH, DH);
    nkb[3] = load_b_frag(k + (size_t)(j1 + 16) * DH + 32, DH);
#pragma unroll
    for (int t = 0; t < 4; ++t)
      nvf[t] = load_b_frag(v + (size_t)(t * 16) * NPIX + j1, NPIX);

    attn_step(aq0, aq1, kb, vf, acc, mrow, lrow, pt, nl, hi);

#pragma unroll
    for (int t = 0; t < 4; ++t) { kb[t] = nkb[t]; vf[t] = nvf[t]; }
  }
  attn_step(aq0, aq1, kb, vf, acc, mrow, lrow, pt, nl, hi);  // last j-step

  // --- normalize and store [n][c] ---
#pragma unroll
  for (int r = 0; r < 8; ++r) {
    int row = i0 + r + 8 * hi;
    float inv = 1.0f / lrow[r];
#pragma unroll
    for (int t = 0; t < 4; ++t)
      ao[((size_t)(b * NPIX + row)) * CDIM + h * DH + t * 16 + nl] =
          (__bf16)(acc[t][r] * inv);
  }
}

// ---------------------------------------------------------------------------
// Kernel 4: output projection + bias + residual, 32x64 tile, pipelined K loop.
// ---------------------------------------------------------------------------
__global__ __launch_bounds__(256) void out_gemm_kernel(
    const __bf16* __restrict__ w, const __bf16* __restrict__ ao,
    const float* __restrict__ bias, const float* __restrict__ x,
    float* __restrict__ out) {
  const int wave = threadIdx.x >> 5, lane = threadIdx.x & 31;
  const int tile = blockIdx.x * 8 + wave;        // 16 * 72 = 1152 tiles
  const int mt = tile % (CDIM / 32);
  const int nt = tile / (CDIM / 32);
  const int m0 = mt * 32, n0 = nt * 64;

  const __bf16* arow0 = w + (size_t)m0 * CDIM;
  const __bf16* arow1 = w + (size_t)(m0 + 16) * CDIM;

  v8f acc[2][4];
#pragma unroll
  for (int mi = 0; mi < 2; ++mi)
#pragma unroll
    for (int t = 0; t < 4; ++t) acc[mi][t] = v8f_zero();

  v16bf a_cur[2], b_cur[4];
  a_cur[0] = load_a_frag(arow0, CDIM);
  a_cur[1] = load_a_frag(arow1, CDIM);
#pragma unroll
  for (int t = 0; t < 4; ++t)
    b_cur[t] = load_b_frag(ao + (size_t)(n0 + 16 * t) * CDIM, CDIM);

#pragma unroll 2
  for (int k0 = 32; k0 < CDIM; k0 += 32) {
    if (k0 + 32 < CDIM) {
      __builtin_prefetch(arow0 + k0 + 32, 0, 3);
      __builtin_prefetch(ao + (size_t)n0 * CDIM + k0 + 32, 0, 3);
    }
    v16bf a_nxt[2], b_nxt[4];
    a_nxt[0] = load_a_frag(arow0 + k0, CDIM);
    a_nxt[1] = load_a_frag(arow1 + k0, CDIM);
#pragma unroll
    for (int t = 0; t < 4; ++t)
      b_nxt[t] = load_b_frag(ao + (size_t)(n0 + 16 * t) * CDIM + k0, CDIM);
#pragma unroll
    for (int t = 0; t < 4; ++t) {
      acc[0][t] = wmma_bf16(a_cur[0], b_cur[t], acc[0][t]);
      acc[1][t] = wmma_bf16(a_cur[1], b_cur[t], acc[1][t]);
    }
    a_cur[0] = a_nxt[0]; a_cur[1] = a_nxt[1];
#pragma unroll
    for (int t = 0; t < 4; ++t) b_cur[t] = b_nxt[t];
  }
#pragma unroll
  for (int t = 0; t < 4; ++t) {
    acc[0][t] = wmma_bf16(a_cur[0], b_cur[t], acc[0][t]);
    acc[1][t] = wmma_bf16(a_cur[1], b_cur[t], acc[1][t]);
  }

  const int nl = lane & 15, hi = lane >> 4;
#pragma unroll
  for (int mi = 0; mi < 2; ++mi) {
#pragma unroll
    for (int t = 0; t < 4; ++t) {
#pragma unroll
      for (int r = 0; r < 8; ++r) {
        int o = m0 + mi * 16 + r + 8 * hi;
        int ngl = n0 + 16 * t + nl;
        int b = ngl / NPIX, n = ngl % NPIX;
        size_t idx = ((size_t)(b * CDIM + o)) * NPIX + n;
        out[idx] = acc[mi][t][r] + bias[o] + x[idx];
      }
    }
  }
}

// ---------------------------------------------------------------------------
// Host launcher
// ---------------------------------------------------------------------------
extern "C" void kernel_launch(void* const* d_in, const int* in_sizes, int n_in,
                              void* d_out, int out_size, void* d_ws, size_t ws_size,
                              hipStream_t stream) {
  const float* x    = (const float*)d_in[0];   // [2,512,48,48]
  const float* g    = (const float*)d_in[1];   // [1,512,1,1]
  const float* wqkv = (const float*)d_in[2];   // [1536,512]
  const float* wout = (const float*)d_in[3];   // [512,512]
  const float* bout = (const float*)d_in[4];   // [512]
  float* out = (float*)d_out;

  char* ws = (char*)d_ws;
  size_t off = 0;
  auto take = [&](size_t bytes) {
    char* p = ws + off;
    off += (bytes + 255) & ~(size_t)255;
    return p;
  };
  __bf16* wqkv_bf = (__bf16*)take((size_t)OQKV * CDIM * 2);
  __bf16* wout_bf = (__bf16*)take((size_t)CDIM * CDIM * 2);
  __bf16* xnt     = (__bf16*)take((size_t)NG * CDIM * 2);
  __bf16* qt      = (__bf16*)take((size_t)NB * HEADS * NPIX * DH * 2);
  __bf16* kt      = (__bf16*)take((size_t)NB * HEADS * NPIX * DH * 2);
  __bf16* vb      = (__bf16*)take((size_t)NB * HEADS * DH * NPIX * 2);
  __bf16* ao      = (__bf16*)take((size_t)NG * CDIM * 2);
  (void)ws_size; (void)in_sizes; (void)n_in; (void)out_size;

  cvt_weights_kernel<<<(OQKV * CDIM + 255) / 256, 256, 0, stream>>>(wqkv, wout,
                                                                    wqkv_bf, wout_bf);
  rmsnorm_kernel<<<(NG + 255) / 256, 256, 0, stream>>>(x, g, xnt);
  qkv_gemm_kernel<<<(OQKV / 32) * (NG / 64) / 8, 256, 0, stream>>>(wqkv_bf, xnt,
                                                                   qt, kt, vb);
  attn_kernel<<<dim3((NPIX / 16) / 4, NB * HEADS), 128, 0, stream>>>(qt, kt, vb, ao);
  out_gemm_kernel<<<(CDIM / 32) * (NG / 64) / 8, 256, 0, stream>>>(wout_bf, ao,
                                                                   bout, x, out);
}
